// Head_9259949490470
// MI455X (gfx1250) — matrix operand order repaired and checked by reading, not compiled, not run on hardware
//
#include <hip/hip_runtime.h>
#include <hip/hip_bf16.h>

// Shapes from the reference
#define Bb 8
#define Tt 2048
#define Cc 1024
#define Hh 64
#define ROWS (Bb * Tt)          // 16384 flattened (b,t) rows
#define NELEM ((size_t)ROWS * Hh)
#define WT_ELEMS ((size_t)Hh * Cc)          // one transposed weight matrix

typedef __attribute__((ext_vector_type(16))) __bf16 v16bf;
typedef __attribute__((ext_vector_type(8)))  __bf16 v8bf;
typedef __attribute__((ext_vector_type(8)))  float  v8f;

// ---------------------------------------------------------------------------
// Kernel 0: one-shot weight transpose + bf16 convert.
//   Wt[w][h][c] = bf16( W_w[c][h] )       (3 x 64 x 1024, 384 KB total)
// In [h, c] layout a WMMA B-operand lane reads 16 contiguous bf16 (one 32B
// vector load) instead of 16 stride-H scalar loads + converts.
// ---------------------------------------------------------------------------
__global__ void __launch_bounds__(256)
wt_kernel(const float* __restrict__ Wk,
          const float* __restrict__ Wq,
          const float* __restrict__ Wv,
          __bf16* __restrict__ wt)
{
    const int w = blockIdx.x;               // 0..2  (K, Q, V)
    const int h = blockIdx.y;               // 0..63
    const float* W = (w == 0) ? Wk : (w == 1) ? Wq : Wv;
    __bf16* dst = wt + (size_t)w * WT_ELEMS + (size_t)h * Cc;
    for (int c = threadIdx.x; c < Cc; c += 256)
        dst[c] = (__bf16)W[(size_t)c * Hh + h];
}

// ---------------------------------------------------------------------------
// Kernel 1: fused Q/K/V projections.  out[m, h] = sum_c x[m, c] * W[c, h]
// One wave per 16-row tile; x tile read ONCE for all three weight matrices
// (12 live f32 accumulators). B operands are single v16bf loads from Wt.
//
// bf16 A as loaded (16x32):    lane row = lane%16; elem i -> K=(hl?16:0)+i
//                              (hl = lane>=16; contiguous k order, same as B)
// bf16 B as loaded (32x16):    lane col = lane%16; elem i -> K=(hl?16:0)+i
// C/D f32 (16x16):             lane col = lane%16; vgpr r -> row = r + 8*hl
//
// Workspace (bf16): K[row,h] | Q[row,h] | V^T[h,row] | Wt[3][h][c]
// ---------------------------------------------------------------------------
__global__ void __launch_bounds__(32)
qkv_proj_kernel(const float* __restrict__ x,
                const __bf16* __restrict__ wt,
                __bf16* __restrict__ ws)
{
    const int g    = blockIdx.x;            // row-tile index, 0..1023
    const int lane = threadIdx.x & 31;
    const int hl   = lane >> 4;
    const int ln   = lane & 15;

    v8f acc[3][4];
    #pragma unroll
    for (int w = 0; w < 3; ++w)
        #pragma unroll
        for (int nt = 0; nt < 4; ++nt)
            #pragma unroll
            for (int r = 0; r < 8; ++r) acc[w][nt][r] = 0.0f;

    const int arow = g * 16 + ln;           // A-operand row for this lane
    for (int c0 = 0; c0 < Cc; c0 += 32) {
        // A tile: x[arow, c0 .. c0+31], 4 x b128 loads, bf16 A-layout
        const float* xr = x + (size_t)arow * Cc + c0 + (hl ? 16 : 0);
        const float4 x0 = *(const float4*)(xr);
        const float4 x1 = *(const float4*)(xr + 4);
        const float4 x2 = *(const float4*)(xr + 8);
        const float4 x3 = *(const float4*)(xr + 12);
        v16bf a;
        a[0]  = (__bf16)x0.x; a[1]  = (__bf16)x0.y; a[2]  = (__bf16)x0.z; a[3]  = (__bf16)x0.w;
        a[4]  = (__bf16)x1.x; a[5]  = (__bf16)x1.y; a[6]  = (__bf16)x1.z; a[7]  = (__bf16)x1.w;
        a[8]  = (__bf16)x2.x; a[9]  = (__bf16)x2.y; a[10] = (__bf16)x2.z; a[11] = (__bf16)x2.w;
        a[12] = (__bf16)x3.x; a[13] = (__bf16)x3.y; a[14] = (__bf16)x3.z; a[15] = (__bf16)x3.w;

        #pragma unroll
        for (int w = 0; w < 3; ++w) {
            #pragma unroll
            for (int nt = 0; nt < 4; ++nt) {
                // B tile: Wt[w][nt*16+ln][c0+K] -> one aligned 32B v16bf load
                const int h = nt * 16 + ln;
                const v16bf b = *(const v16bf*)(
                    wt + (size_t)w * WT_ELEMS + (size_t)h * Cc
                       + c0 + (hl ? 16 : 0));
                acc[w][nt] = __builtin_amdgcn_wmma_f32_16x16x32_bf16(
                    false, a, false, b, (short)0, acc[w][nt], false, false);
            }
        }
    }

    // K (w=0) and Q (w=1): row-major [row, h]
    #pragma unroll
    for (int w = 0; w < 2; ++w) {
        __bf16* dst = ws + (size_t)w * NELEM;
        #pragma unroll
        for (int nt = 0; nt < 4; ++nt)
            #pragma unroll
            for (int r = 0; r < 8; ++r) {
                const int orow = g * 16 + r + 8 * hl;
                dst[(size_t)orow * Hh + nt * 16 + ln] = (__bf16)acc[w][nt][r];
            }
    }
    // V (w=2): transposed [h, row]; 8 contiguous bf16 per lane -> vector store
    {
        __bf16* vt = ws + 2 * NELEM;
        #pragma unroll
        for (int nt = 0; nt < 4; ++nt) {
            v8bf pk;
            #pragma unroll
            for (int r = 0; r < 8; ++r) pk[r] = (__bf16)acc[2][nt][r];
            const int h = nt * 16 + ln;
            *(v8bf*)(vt + (size_t)h * ROWS + g * 16 + 8 * hl) = pk;
        }
    }
}

// ---------------------------------------------------------------------------
// Kernel 2: flash attention over one 16-row query tile per wave.
// scores = (Q K^T) * H^-0.5 with causal mask, online softmax, out = P V.
// ---------------------------------------------------------------------------
__global__ void __launch_bounds__(32)
attn_kernel(const __bf16* __restrict__ ws, float* __restrict__ out)
{
    const __bf16* kbf = ws;                 // [row, h]
    const __bf16* qbf = ws + NELEM;         // [row, h]
    const __bf16* vtb = ws + 2 * NELEM;     // [h, row]  (transposed V)

    const int g    = blockIdx.x;            // 0..1023
    const int b    = g >> 7;                // batch
    const int tt   = g & 127;               // query tile within batch
    const int base = b * Tt;                // flat row base of this batch
    const int t0   = tt * 16;               // local query row offset

    const int lane = threadIdx.x & 31;
    const int hl   = lane >> 4;
    const int ln   = lane & 15;

    // Load Q tile once in A-layout: two K=32 chunks covering H=64
    v16bf aq[2];
    {
        const __bf16* qr = qbf + (size_t)(base + t0 + ln) * Hh;
        #pragma unroll
        for (int ch = 0; ch < 2; ++ch) {
            const v8bf lo = *(const v8bf*)(qr + ch * 32 + (hl ? 16 : 0));
            const v8bf hi = *(const v8bf*)(qr + ch * 32 + 8 + (hl ? 16 : 0));
            #pragma unroll
            for (int i = 0; i < 8; ++i) { aq[ch][i] = lo[i]; aq[ch][i + 8] = hi[i]; }
        }
    }

    float m[8], l[8];
    v8f acc[4];
    #pragma unroll
    for (int r = 0; r < 8; ++r) { m[r] = -1e30f; l[r] = 0.0f; }
    #pragma unroll
    for (int nt = 0; nt < 4; ++nt)
        #pragma unroll
        for (int r = 0; r < 8; ++r) acc[nt][r] = 0.0f;

    __shared__ __bf16 pl[16 * 16];          // P-tile bounce buffer (512 B)

    for (int st = 0; st <= tt; ++st) {
        const int s0 = st * 16;

        // Prefetch next key tile into cache (CDNA5 global_prefetch path)
        if (st < tt)
            __builtin_prefetch(kbf + (size_t)(base + s0 + 16 + ln) * Hh, 0, 3);

        // ---- scores tile S = Q K^T  (16x16, K=64): 2 WMMAs ----
        v8f s;
        #pragma unroll
        for (int r = 0; r < 8; ++r) s[r] = 0.0f;
        #pragma unroll
        for (int ch = 0; ch < 2; ++ch) {
            // B-layout: col = key row s0+ln; 16 contiguous bf16 = one 32B load
            const v16bf bk = *(const v16bf*)(
                kbf + (size_t)(base + s0 + ln) * Hh + ch * 32 + (hl ? 16 : 0));
            s = __builtin_amdgcn_wmma_f32_16x16x32_bf16(
                false, aq[ch], false, bk, (short)0, s, false, false);
        }

        // ---- scale + causal mask (diagonal tile only) ----
        float sc[8];
        #pragma unroll
        for (int r = 0; r < 8; ++r) {
            float v = s[r] * 0.125f;        // H^-0.5 = 1/8
            if (st == tt && ln > (r + 8 * hl)) v = -1e30f;
            sc[r] = v;
        }

        // ---- online softmax: row reductions within 16-lane half ----
        float mnew[8], corr[8], p[8];
        #pragma unroll
        for (int r = 0; r < 8; ++r) {
            float v = sc[r];
            v = fmaxf(v, __shfl_xor(v, 1, 32));
            v = fmaxf(v, __shfl_xor(v, 2, 32));
            v = fmaxf(v, __shfl_xor(v, 4, 32));
            v = fmaxf(v, __shfl_xor(v, 8, 32));
            mnew[r] = fmaxf(m[r], v);
            corr[r] = __expf(m[r] - mnew[r]);
            p[r]    = __expf(sc[r] - mnew[r]);
        }
        #pragma unroll
        for (int r = 0; r < 8; ++r) {
            float v = p[r];
            v += __shfl_xor(v, 1, 32);
            v += __shfl_xor(v, 2, 32);
            v += __shfl_xor(v, 4, 32);
            v += __shfl_xor(v, 8, 32);
            l[r] = l[r] * corr[r] + v;
            m[r] = mnew[r];
        }
        #pragma unroll
        for (int nt = 0; nt < 4; ++nt)
            #pragma unroll
            for (int r = 0; r < 8; ++r) acc[nt][r] *= corr[r];

        // ---- transpose P (D-layout) -> A-layout via LDS ----
        __syncthreads();
        #pragma unroll
        for (int r = 0; r < 8; ++r)
            pl[(r + 8 * hl) * 16 + ln] = (__bf16)p[r];
        __syncthreads();

        v16bf ap;   // A 16x32: K=0..15 = P columns, K>=16 zero-padded
        #pragma unroll
        for (int i = 0; i < 8; ++i) {
            ap[i]     = hl ? (__bf16)0.0f : pl[ln * 16 + i];
            ap[i + 8] = hl ? (__bf16)0.0f : pl[ln * 16 + 8 + i];
        }

        // ---- out += P V  (4 N-tiles over H=64): 4 WMMAs ----
        #pragma unroll
        for (int nt = 0; nt < 4; ++nt) {
            v16bf bv;   // B-layout: elem i -> K=(hl?16:0)+i; K>=16 unused (A==0)
            #pragma unroll
            for (int i = 0; i < 16; ++i) bv[i] = (__bf16)0.0f;
            if (hl == 0) {
                // transposed V: 16 contiguous bf16 in the sequence dim
                bv = *(const v16bf*)(
                    vtb + (size_t)(nt * 16 + ln) * ROWS + base + s0);
            }
            acc[nt] = __builtin_amdgcn_wmma_f32_16x16x32_bf16(
                false, ap, false, bv, (short)0, acc[nt], false, false);
        }
    }

    // ---- epilogue: divide by softmax denominator, store fp32 ----
    #pragma unroll
    for (int nt = 0; nt < 4; ++nt)
        #pragma unroll
        for (int r = 0; r < 8; ++r) {
            const int orow = base + t0 + r + 8 * hl;
            out[(size_t)orow * Hh + nt * 16 + ln] = acc[nt][r] / l[r];
        }
}

// ---------------------------------------------------------------------------
extern "C" void kernel_launch(void* const* d_in, const int* in_sizes, int n_in,
                              void* d_out, int out_size, void* d_ws, size_t ws_size,
                              hipStream_t stream)
{
    const float* x  = (const float*)d_in[0];
    const float* Wk = (const float*)d_in[1];
    const float* Wq = (const float*)d_in[2];
    const float* Wv = (const float*)d_in[3];
    float* out      = (float*)d_out;
    __bf16* ws      = (__bf16*)d_ws;    // 3*NELEM + 3*WT_ELEMS bf16 ~ 6.4 MB
    __bf16* wt      = ws + 3 * NELEM;   // transposed bf16 weights

    dim3 gridW(3, Hh);                  // one-shot weight transpose (384 KB)
    wt_kernel<<<gridW, 256, 0, stream>>>(Wk, Wq, Wv, wt);

    dim3 gridP(ROWS / 16);              // 1024 fused projection tiles
    qkv_proj_kernel<<<gridP, 32, 0, stream>>>(x, wt, ws);

    dim3 gridA(ROWS / 16);              // 1024 query tiles
    attn_kernel<<<gridA, 32, 0, stream>>>(ws, out);
}
